// DynamicGraphAttention_12987981103577
// MI455X (gfx1250) — compile-verified, matrix operand order checked by reading, not run on hardware
//
#include <hip/hip_runtime.h>

#define BB 16
#define NN 2048
#define CC 128

typedef __attribute__((ext_vector_type(16))) __bf16 v16bf;
typedef __attribute__((ext_vector_type(8)))  float  v8f;

union Frag16 {
    v16bf bf;
    uint4 q[2];
};
union AccU {
    v8f   v;
    float f[8];
};

__device__ __forceinline__ unsigned short f2bf(float x) {
    unsigned int u = __float_as_uint(x);
    u += 0x7FFFu + ((u >> 16) & 1u);   // round-to-nearest-even
    return (unsigned short)(u >> 16);
}

// ---------------------------------------------------------------------------
// Kernel 1: W_sym = 0.5 * (Wq^T Wk + Wk^T Wq) * (1 / (sqrt(C)*TAU))
// grid(128) x block(128); c1 = blockIdx, c2 = threadIdx
// ---------------------------------------------------------------------------
__global__ void wsym_kernel(const float* __restrict__ Wq,
                            const float* __restrict__ Wk,
                            float* __restrict__ Ws) {
    const int c2 = threadIdx.x;
    const int c1 = blockIdx.x;
    float s = 0.f;
    for (int d = 0; d < CC; ++d) {
        float wq1 = Wq[d * CC + c1];
        float wk1 = Wk[d * CC + c1];
        float wq2 = Wq[d * CC + c2];
        float wk2 = Wk[d * CC + c2];
        s += wq1 * wk2 + wk1 * wq2;
    }
    const float scale = 0.5f / (sqrtf((float)CC) * 1.5f);  // 0.5 * 1/(sqrt(C)*TAU)
    Ws[c1 * CC + c2] = s * scale;
}

// ---------------------------------------------------------------------------
// Kernel 2: G = H @ W_sym (bf16 out), Hb = bf16(H).
// One block per 16 rows, 128 threads (one output column each).
// ---------------------------------------------------------------------------
__global__ void proj_kernel(const float* __restrict__ H,
                            const float* __restrict__ Ws,
                            unsigned short* __restrict__ Gb,
                            unsigned short* __restrict__ Hb) {
    __shared__ float h[16][CC];
    const int d = threadIdx.x;
    const long long row0 = (long long)blockIdx.x * 16;

    for (int r = 0; r < 16; ++r) {
        float v = H[(row0 + r) * CC + d];
        h[r][d] = v;
        Hb[(row0 + r) * CC + d] = f2bf(v);
    }
    __syncthreads();

    float acc[16];
#pragma unroll
    for (int r = 0; r < 16; ++r) acc[r] = 0.f;

    for (int c = 0; c < CC; ++c) {
        float w = Ws[c * CC + d];
#pragma unroll
        for (int r = 0; r < 16; ++r) acc[r] += h[r][c] * w;
    }
#pragma unroll
    for (int r = 0; r < 16; ++r) Gb[(row0 + r) * CC + d] = f2bf(acc[r]);
}

// ---------------------------------------------------------------------------
// Kernel 3: logits = G @ H^T (+bias/masks), fused row softmax, single pass.
// Block = 512 threads = 16 waves; each block owns (b, 16-row stripe).
// Each wave computes 8 consecutive 16x16 j-tiles via v_wmma_f32_16x16x32_bf16,
// keeps logits in registers, block-wide softmax via shfl + 1KB LDS.
// ---------------------------------------------------------------------------
__global__ __launch_bounds__(512)
void attn_kernel(const unsigned short* __restrict__ Gb,
                 const unsigned short* __restrict__ Hb,
                 const float* __restrict__ A_stat,
                 const float* __restrict__ M_mask,
                 float* __restrict__ out) {
    constexpr int NT = 8;                    // j-tiles per wave
    const int tid   = threadIdx.x;
    const int lane  = tid & 31;
    const int wave  = tid >> 5;              // 0..15
    const int n     = lane & 15;             // column-in-tile / row-in-tile
    const int khalf = (lane >> 4) * 8;       // K-half (elements) per ISA layout

    const int bid = blockIdx.x;
    const int b   = bid >> 7;                // N/16 == 128 stripes per batch
    const int i0  = (bid & 127) << 4;

    __shared__ float redbuf[16][16];
    __shared__ float rowstat[16];

    // Preload A fragments (rows i0+n, full K=128): 4 chunks x 32 bf16
    Frag16 afrag[4];
    const uint4* Arow = (const uint4*)(Gb + ((size_t)b * NN + i0 + n) * CC);
#pragma unroll
    for (int kk = 0; kk < 4; ++kk) {
        afrag[kk].q[0] = Arow[(kk * 32 + khalf) >> 3];
        afrag[kk].q[1] = Arow[(kk * 32 + khalf + 16) >> 3];
    }

    AccU acc[NT];
    const int jbase = wave * (NT * 16);

#pragma unroll
    for (int t = 0; t < NT; ++t) {
        const int j0 = jbase + t * 16;
        const uint4* Brow = (const uint4*)(Hb + ((size_t)b * NN + j0 + n) * CC);
        v8f c = {0.f, 0.f, 0.f, 0.f, 0.f, 0.f, 0.f, 0.f};
#pragma unroll
        for (int kk = 0; kk < 4; ++kk) {
            Frag16 bf;
            bf.q[0] = Brow[(kk * 32 + khalf) >> 3];
            bf.q[1] = Brow[(kk * 32 + khalf + 16) >> 3];
            c = __builtin_amdgcn_wmma_f32_16x16x32_bf16(
                    false, afrag[kk].bf, false, bf.bf, (short)0, c, false, false);
        }
        acc[t].v = c;
    }

    const float NEG = -__builtin_inff();

    // Bias + masks + per-lane row max. C-layout: vgpr r -> row r (+8 if lane>=16),
    // lanes n=0..15 -> columns.
    float pmax[8];
#pragma unroll
    for (int r = 0; r < 8; ++r) pmax[r] = NEG;

#pragma unroll
    for (int t = 0; t < NT; ++t) {
        const int col = jbase + t * 16 + n;
#pragma unroll
        for (int r = 0; r < 8; ++r) {
            const int i = i0 + r + khalf;
            const size_t idx = (size_t)i * NN + col;
            float v = acc[t].f[r];
            v = (i == col) ? NEG : (v + A_stat[idx]);      // eye -> -inf; +ETA*A_bias
            if (M_mask[idx] <= 0.f) v = NEG;               // external mask
            acc[t].f[r] = v;
            pmax[r] = fmaxf(pmax[r], v);
        }
    }

    // Reduce max across the 16 lanes of each half-wave (xor masks < 16).
#pragma unroll
    for (int r = 0; r < 8; ++r)
#pragma unroll
        for (int off = 1; off < 16; off <<= 1)
            pmax[r] = fmaxf(pmax[r], __shfl_xor(pmax[r], off, 32));

#pragma unroll
    for (int r = 0; r < 8; ++r)
        if ((lane & 15) == r) redbuf[wave][r + khalf] = pmax[r];
    __syncthreads();

    if (tid < 16) {
        float m = NEG;
        for (int w = 0; w < 16; ++w) m = fmaxf(m, redbuf[w][tid]);
        rowstat[tid] = m;
    }
    __syncthreads();

    float rowmax[8];
#pragma unroll
    for (int r = 0; r < 8; ++r) rowmax[r] = rowstat[r + khalf];

    // exp + per-lane partial sums (logits overwritten with exp values)
    float psum[8];
#pragma unroll
    for (int r = 0; r < 8; ++r) psum[r] = 0.f;
#pragma unroll
    for (int t = 0; t < NT; ++t)
#pragma unroll
        for (int r = 0; r < 8; ++r) {
            float e = __expf(acc[t].f[r] - rowmax[r]);
            acc[t].f[r] = e;
            psum[r] += e;
        }
#pragma unroll
    for (int r = 0; r < 8; ++r)
#pragma unroll
        for (int off = 1; off < 16; off <<= 1)
            psum[r] += __shfl_xor(psum[r], off, 32);

#pragma unroll
    for (int r = 0; r < 8; ++r)
        if ((lane & 15) == r) redbuf[wave][r + khalf] = psum[r];
    __syncthreads();

    if (tid < 16) {
        float s = 0.f;
        for (int w = 0; w < 16; ++w) s += redbuf[w][tid];
        rowstat[tid] = 1.0f / s;
    }
    __syncthreads();

    float rinv[8];
#pragma unroll
    for (int r = 0; r < 8; ++r) rinv[r] = rowstat[r + khalf];

    // Streamed output write (mandatory 256 MB; this bounds the kernel)
#pragma unroll
    for (int t = 0; t < NT; ++t) {
        const int col = jbase + t * 16 + n;
#pragma unroll
        for (int r = 0; r < 8; ++r) {
            const int i = i0 + r + khalf;
            out[((size_t)b * NN + i) * NN + col] = acc[t].f[r] * rinv[r];
        }
    }
}

// ---------------------------------------------------------------------------
extern "C" void kernel_launch(void* const* d_in, const int* in_sizes, int n_in,
                              void* d_out, int out_size, void* d_ws, size_t ws_size,
                              hipStream_t stream) {
    const float* H      = (const float*)d_in[0];  // [B,N,C]
    const float* A_stat = (const float*)d_in[1];  // [N,N]
    const float* M_mask = (const float*)d_in[2];  // [N,N]
    const float* Wq     = (const float*)d_in[3];  // [C,C]
    const float* Wk     = (const float*)d_in[4];  // [C,C]
    float* out = (float*)d_out;                   // [B,N,N]

    char* ws = (char*)d_ws;
    float*          Ws = (float*)ws;                                   // 64 KB
    unsigned short* Gb = (unsigned short*)(ws + (64 << 10));           // 8 MB
    unsigned short* Hb = (unsigned short*)(ws + (64 << 10) +
                                           (size_t)BB * NN * CC * 2);  // 8 MB

    wsym_kernel<<<CC, CC, 0, stream>>>(Wq, Wk, Ws);
    proj_kernel<<<(BB * NN) / 16, CC, 0, stream>>>(H, Ws, Gb, Hb);
    attn_kernel<<<BB * (NN / 16), 512, 0, stream>>>(Gb, Hb, A_stat, M_mask, out);
}
